// TransducerAcousticModle_81578608820616
// MI455X (gfx1250) — matrix operand order repaired and checked by reading, not compiled, not run on hardware
//
#include <hip/hip_runtime.h>
#include <hip/hip_fp16.h>

typedef __attribute__((ext_vector_type(16))) _Float16 v16h;
typedef __attribute__((ext_vector_type(8)))  _Float16 v8h;
typedef __attribute__((ext_vector_type(8)))  float    v8f;

#define B_  4
#define T_  256
#define U_  64
#define D_  512
#define K_  512    // INNER
#define V_  2048   // VOCAB

#define HPAD 8
#define HROW (K_ + HPAD)   // 520 halves per LDS row (16B-aligned rows, bank spread)

// ---------------- prep kernels (tiny) ----------------

__global__ void cvt_f32_to_f16(const float* __restrict__ src,
                               _Float16* __restrict__ dst, int n) {
  int i = blockIdx.x * blockDim.x + threadIdx.x;
  if (i < n) dst[i] = (_Float16)src[i];
}

// Repack W (row-major [k][n], row stride ld, f32) into WMMA B-fragment order:
//   frag[tile][kstep][lane][0..15]; lane<16 holds column n=tile*16+lane,
//   K = kstep*32 + {0..7, 16..23}; lane>=16 same column, K = kstep*32 + {8..15, 24..31}.
__global__ void frag_w(const float* __restrict__ src, int ld,
                       _Float16* __restrict__ dst, int ntiles, int ksteps) {
  int idx = blockIdx.x * blockDim.x + threadIdx.x;
  int total = ntiles * ksteps * 32;
  if (idx >= total) return;
  int lane = idx & 31;
  int ks   = (idx >> 5) % ksteps;
  int nt   = idx / (32 * ksteps);
  int n     = nt * 16 + (lane & 15);
  int kbase = ks * 32 + ((lane >> 4) << 3);
  v16h vv;
#pragma unroll
  for (int j = 0; j < 16; ++j)
    vv[j] = (_Float16)src[(long)(kbase + j + ((j >> 3) << 3)) * ld + n];
  ((v16h*)dst)[idx] = vv;
}

// ---------------- projection GEMM: (M x 512) = A_f16(M x 512) @ Wfrag ----------------
__global__ __launch_bounds__(256)
void proj_gemm(const _Float16* __restrict__ A,      // row-major, stride K_
               const _Float16* __restrict__ Bfrag,  // 32 ntiles x 16 ksteps
               float* __restrict__ out) {           // row-major, stride 512
  int wave = threadIdx.x >> 5;
  int lane = threadIdx.x & 31;
  int m0   = blockIdx.x * 16;
  int row   = lane & 15;
  int khalf = (lane >> 4) ? 8 : 0;
  v8f acc[4] = {};
  const _Float16* arow = A + (long)(m0 + row) * K_;
#pragma unroll 1
  for (int s = 0; s < 16; ++s) {
    int k0 = s * 32 + khalf;
    union { v16h v; v8h h[2]; } a;
    a.h[0] = *(const v8h*)(arow + k0);
    a.h[1] = *(const v8h*)(arow + k0 + 16);
#pragma unroll
    for (int nt = 0; nt < 4; ++nt) {
      int tile = wave * 4 + nt;
      v16h bf = ((const v16h*)Bfrag)[(tile * 16 + s) * 32 + lane];
      acc[nt] = __builtin_amdgcn_wmma_f32_16x16x32_f16(
          false, a.v, false, bf, (short)0, acc[nt], false, false);
    }
  }
  int mbase = (lane >> 4) * 8;
#pragma unroll
  for (int nt = 0; nt < 4; ++nt) {
    int col = (wave * 4 + nt) * 16 + (lane & 15);
#pragma unroll
    for (int j = 0; j < 8; ++j)
      out[(long)(m0 + mbase + j) * 512 + col] = acc[nt][j];
  }
}

// ---------------- fused joint kernel ----------------
// grid = (B*T/2) * 4 N-groups; block = 512 (16 waves).
// Block tile: M = 128 rows (2 t-values x 64 u), N = 512 cols.
// Each wave: 128x32 output (8 M-tiles x 2 N-tiles = 16 f32 accumulators).
__global__ __launch_bounds__(512)
void joint_kernel(const float* __restrict__ enc_proj,  // (B*T, 512)
                  const float* __restrict__ dec_proj,  // (B*U, 512)
                  const float* __restrict__ b1,        // (512)
                  const _Float16* __restrict__ W2frag, // 128 ntiles x 16 ksteps
                  const float* __restrict__ b2,        // (2048)
                  float* __restrict__ out) {           // (B,T,U,V)
  __shared__ float    s_eb[2 * K_];
  __shared__ _Float16 s_h[128 * HROW];   // ~133 KB of the 320 KB WGP LDS

  int blk = blockIdx.x;
  int ng  = blk & 3;           // N group: 512 cols each
  int mg  = blk >> 2;          // 0..511 : (b, t-pair)
  int bt0 = mg * 2;            // first flattened (b*T+t) row
  int bb  = bt0 >> 8;          // batch index (T = 256)
  int tid = threadIdx.x;

  // phase 0: two enc_proj rows + b1
  s_eb[tid]      = enc_proj[(long)bt0 * K_ + tid] + b1[tid];
  s_eb[K_ + tid] = enc_proj[(long)(bt0 + 1) * K_ + tid] + b1[tid];
  __syncthreads();

  // phase 1: 128x512 tanh tile -> LDS f16; dec_proj read once per block
  const float* dpr = dec_proj + (long)bb * U_ * K_;
  float eb0 = s_eb[tid], eb1 = s_eb[K_ + tid];
#pragma unroll 1
  for (int r = 0; r < 64; ++r) {
    float d = dpr[(long)r * K_ + tid];
    s_h[r * HROW + tid]        = (_Float16)tanhf(eb0 + d);
    s_h[(64 + r) * HROW + tid] = (_Float16)tanhf(eb1 + d);
  }
  __syncthreads();

  // phase 2: K loop, 16 WMMAs per step per wave
  int wave  = tid >> 5;
  int lane  = tid & 31;
  int row   = lane & 15;
  int khalf = (lane >> 4) ? 8 : 0;
  int tileN0 = ng * 32 + wave * 2;     // global n-tile base (of 128)

  v8f acc[8][2] = {};
#pragma unroll 1
  for (int s = 0; s < 16; ++s) {
    int k0 = s * 32 + khalf;
    const v16h* bp = (const v16h*)W2frag + ((tileN0 * 16 + s) * 32 + lane);
    v16h bf0 = bp[0];                  // coalesced 32B/lane, L2-hot
    v16h bf1 = bp[16 * 32];
#pragma unroll
    for (int mt = 0; mt < 8; ++mt) {
      const _Float16* hr = s_h + (mt * 16 + row) * HROW;
      union { v16h v; v8h h[2]; } a;
      a.h[0] = *(const v8h*)(hr + k0);        // ds_load_b128
      a.h[1] = *(const v8h*)(hr + k0 + 16);   // ds_load_b128
      acc[mt][0] = __builtin_amdgcn_wmma_f32_16x16x32_f16(
          false, a.v, false, bf0, (short)0, acc[mt][0], false, false);
      acc[mt][1] = __builtin_amdgcn_wmma_f32_16x16x32_f16(
          false, a.v, false, bf1, (short)0, acc[mt][1], false, false);
    }
  }

  // epilogue: + b2, store f32
  int mbase = (lane >> 4) * 8;
#pragma unroll
  for (int nt = 0; nt < 2; ++nt) {
    int col = (tileN0 + nt) * 16 + (lane & 15);
    float bias = b2[col];
#pragma unroll
    for (int mt = 0; mt < 8; ++mt) {
#pragma unroll
      for (int j = 0; j < 8; ++j) {
        int ml = mt * 16 + mbase + j;                  // 0..127
        long o = ((long)(bt0 + (ml >> 6)) * U_ + (ml & 63)) * V_ + col;
        out[o] = acc[mt][nt][j] + bias;
      }
    }
  }
}

// ---------------- launch ----------------

extern "C" void kernel_launch(void* const* d_in, const int* in_sizes, int n_in,
                              void* d_out, int out_size, void* d_ws, size_t ws_size,
                              hipStream_t stream) {
  const float* enc = (const float*)d_in[0];   // (B,T,D)
  const float* dec = (const float*)d_in[1];   // (B,U,D)
  const float* W1  = (const float*)d_in[2];   // (2D, INNER)
  const float* b1  = (const float*)d_in[3];   // (INNER)
  const float* W2  = (const float*)d_in[4];   // (INNER, VOCAB)
  const float* b2  = (const float*)d_in[5];   // (VOCAB)
  float* out = (float*)d_out;

  char* ws = (char*)d_ws;
  size_t off = 0;
  _Float16* enc16 = (_Float16*)(ws + off); off += (size_t)B_ * T_ * D_ * 2;
  _Float16* dec16 = (_Float16*)(ws + off); off += (size_t)B_ * U_ * D_ * 2;
  _Float16* WeF   = (_Float16*)(ws + off); off += (size_t)32 * 16 * 32 * 16 * 2;
  _Float16* WdF   = (_Float16*)(ws + off); off += (size_t)32 * 16 * 32 * 16 * 2;
  _Float16* W2F   = (_Float16*)(ws + off); off += (size_t)128 * 16 * 32 * 16 * 2;
  float*    encp  = (float*)(ws + off);    off += (size_t)B_ * T_ * K_ * 4;
  float*    decp  = (float*)(ws + off);    off += (size_t)B_ * U_ * K_ * 4;

  int n1 = B_ * T_ * D_;
  cvt_f32_to_f16<<<(n1 + 255) / 256, 256, 0, stream>>>(enc, enc16, n1);
  int n2 = B_ * U_ * D_;
  cvt_f32_to_f16<<<(n2 + 255) / 256, 256, 0, stream>>>(dec, dec16, n2);

  int fw1 = 32 * 16 * 32;
  frag_w<<<(fw1 + 255) / 256, 256, 0, stream>>>(W1, K_, WeF, 32, 16);
  frag_w<<<(fw1 + 255) / 256, 256, 0, stream>>>(W1 + (size_t)D_ * K_, K_, WdF, 32, 16);
  int fw2 = 128 * 16 * 32;
  frag_w<<<(fw2 + 255) / 256, 256, 0, stream>>>(W2, V_, W2F, 128, 16);

  proj_gemm<<<(B_ * T_) / 16, 256, 0, stream>>>(enc16, WeF, encp);
  proj_gemm<<<(B_ * U_) / 16, 256, 0, stream>>>(dec16, WdF, decp);

  joint_kernel<<<(B_ * T_ / 2) * 4, 512, 0, stream>>>(encp, decp, b1, W2F, b2, out);
}
